// BiLSTMNet_41059887349884
// MI455X (gfx1250) — compile-verified, matrix operand order checked by reading, not run on hardware
//
#include <hip/hip_runtime.h>
#include <hip/hip_bf16.h>
#include <math.h>

typedef __bf16 bf16;
typedef __bf16 v16bf __attribute__((ext_vector_type(16)));
typedef float  v8f   __attribute__((ext_vector_type(8)));
typedef int    v4i_  __attribute__((ext_vector_type(4)));

#define Tdim 512
#define Bdim 128
#define Edim 200
#define Hdim 200
#define H4   800
#define Pdim 256
#define MLPD 200
#define Cdim 4
#define NTP  52        // N tiles padded so each of 4 waves owns exactly 13
#define PREW 832       // NTP*16 columns in the pre-activation LDS tile
#define KTH  7         // ceil(200/32) k-tiles for the recurrent segment
#define XPAD0 224      // 7*32  (layer0 input, E=200 padded)
#define HOPAD 416      // hidden output stride (2H=400 padded to 13*32)

#if defined(__has_builtin)
#if __has_builtin(__builtin_amdgcn_global_load_async_to_lds_b128) && \
    __has_builtin(__builtin_amdgcn_s_wait_asynccnt)
#define USE_ASYNC_LDS 1
#endif
#endif
#ifndef USE_ASYNC_LDS
#define USE_ASYNC_LDS 0
#endif

#if USE_ASYNC_LDS
typedef __attribute__((address_space(1))) v4i_* gv4p;
typedef __attribute__((address_space(3))) v4i_* lv4p;
#endif

union FragBf { float4 q[2]; v16bf v; };

__device__ __forceinline__ float sigm(float x) { return 1.0f / (1.0f + __expf(-x)); }

__device__ __forceinline__ float fast_tanh(float x) {
#if defined(__has_builtin) && __has_builtin(__builtin_amdgcn_tanhf)
  return __builtin_amdgcn_tanhf(x);
#elif defined(__has_builtin) && __has_builtin(__builtin_amdgcn_tanh_f32)
  return __builtin_amdgcn_tanh_f32(x);
#else
  return tanhf(x);
#endif
}

// ---------------- embedding gather -> padded bf16 x ----------------
__global__ void k_embed(const int* __restrict__ tok, const float* __restrict__ emb,
                        bf16* __restrict__ x0, int total) {
  int i = blockIdx.x * blockDim.x + threadIdx.x;
  if (i >= total) return;
  int e = i % XPAD0;
  int tb = i / XPAD0;
  float v = 0.0f;
  if (e < Edim) {
    int t = tok[tb];
    v = emb[(size_t)t * Edim + e];
  }
  x0[i] = (bf16)v;
}

// ---------------- zero fill (64-bit) ----------------
__global__ void k_zero64(unsigned long long* __restrict__ p, size_t n) {
  size_t i = (size_t)blockIdx.x * blockDim.x + threadIdx.x;
  if (i < n) p[i] = 0ull;
}

// ---- prepack [wih | whh] into per-lane WMMA B-fragment order ----
// layout: [nt(52)][kt][lane(32)][e(16)] bf16
__global__ void k_pack_lstm(const float* __restrict__ wih, const float* __restrict__ whh,
                            int din, int ktx, bf16* __restrict__ out, int total) {
  int i = blockIdx.x * blockDim.x + threadIdx.x;
  if (i >= total) return;
  int e = i & 15;
  int lane = (i >> 4) & 31;
  int rest = i >> 9;                 // nt*KT + kt
  int KT = ktx + KTH;
  int kt = rest % KT;
  int nt = rest / KT;
  int n = nt * 16 + (lane & 15);
  int hs = lane >> 4;
  int kk = e + 8 * hs + (e >= 8 ? 8 : 0);   // ISA 16-bit B fragment K mapping
  float v = 0.0f;
  if (n < H4) {
    if (kt < ktx) {
      int k = kt * 32 + kk;
      if (k < din) v = wih[(size_t)n * din + k];
    } else {
      int k = (kt - ktx) * 32 + kk;
      if (k < Hdim) v = whh[(size_t)n * Hdim + k];
    }
  }
  out[i] = (bf16)v;
}

// ---------------- bias sum ----------------
__global__ void k_bsum(const float* __restrict__ a, const float* __restrict__ b,
                       float* __restrict__ o, int n) {
  int i = blockIdx.x * blockDim.x + threadIdx.x;
  if (i < n) o[i] = a[i] + b[i];
}

// ---- prepack w1 (MLP x 800) into B-fragment order, N padded to 208 ----
__global__ void k_pack_w1(const float* __restrict__ w1, bf16* __restrict__ out, int total) {
  int i = blockIdx.x * blockDim.x + threadIdx.x;
  if (i >= total) return;
  int e = i & 15;
  int lane = (i >> 4) & 31;
  int rest = i >> 9;                 // nt*25 + kt
  int kt = rest % 25;
  int nt = rest / 25;
  int n = nt * 16 + (lane & 15);
  int hs = lane >> 4;
  int kk = e + 8 * hs + (e >= 8 ? 8 : 0);
  int k = kt * 32 + kk;
  float v = (n < MLPD) ? w1[(size_t)n * H4 + k] : 0.0f;
  out[i] = (bf16)v;
}

// ---------------- fused bi-LSTM layer ----------------
// grid = 8 blocks: (batch-slice of 32) x (dir). 128 threads = 4 wave32.
// Per step: pre = [x_t, h_prev] @ Wpack via v_wmma_f32_16x16x32_bf16 with a
// 4-deep software-pipelined weight stream (B reused across two M tiles),
// then elementwise LSTM gates.
template <int KTX>
__global__ __launch_bounds__(128) void k_bilstm(
    const bf16* __restrict__ xin,
    const bf16* __restrict__ wpF, const bf16* __restrict__ wpB,
    const float* __restrict__ bsF, const float* __restrict__ bsB,
    bf16* __restrict__ hout) {
  constexpr int XPAD = KTX * 32;
  constexpr int KT = KTX + KTH;
  __shared__ __align__(16) bf16 xtile[32][XPAD];
  __shared__ __align__(16) bf16 hprev[32][KTH * 32];
  __shared__ __align__(16) float pre[32][PREW];
  __shared__ float cst[32][Hdim];

  const int tid = threadIdx.x;
  const int dir = blockIdx.x & 1;
  const int b0 = (blockIdx.x >> 1) * 32;
  const bf16* wp = dir ? wpB : wpF;
  const float* bs = dir ? bsB : bsF;
  const int wave = tid >> 5;
  const int lane = tid & 31;
  const int col = lane & 15;
  const int hs = lane >> 4;

  for (int i = tid; i < 32 * KTH * 32; i += 128) ((bf16*)hprev)[i] = (bf16)0.0f;
  for (int i = tid; i < 32 * Hdim; i += 128) ((float*)cst)[i] = 0.0f;

  for (int ti = 0; ti < Tdim; ++ti) {
    const int t = dir ? (Tdim - 1 - ti) : ti;

    // stage x_t tile (32 rows x XPAD, contiguous) into LDS
    const bf16* src = xin + ((size_t)t * Bdim + b0) * XPAD;
#if USE_ASYNC_LDS
    {
      char* gp = (char*)(void*)const_cast<bf16*>(src);
      char* lp = (char*)(void*)&xtile[0][0];
      const int bytes = 32 * XPAD * 2;
      for (int i = tid * 16; i < bytes; i += 128 * 16)
        __builtin_amdgcn_global_load_async_to_lds_b128((gv4p)(gp + i), (lv4p)(lp + i), 0, 0);
      __builtin_amdgcn_s_wait_asynccnt(0);
    }
#else
    {
      const float4* s4 = (const float4*)src;
      float4* d4 = (float4*)&xtile[0][0];
      const int n4 = 32 * XPAD / 8;  // 8 bf16 per float4
      for (int i = tid; i < n4; i += 128) d4[i] = s4[i];
    }
#endif
    // prefetch next time-step's x tile into cache while we compute
    if (ti + 1 < Tdim) {
      const bf16* nsrc = xin + ((size_t)(dir ? t - 1 : t + 1) * Bdim + b0) * XPAD;
      for (int i = tid * 64; i < 32 * XPAD; i += 128 * 64)
        __builtin_prefetch(nsrc + i, 0, 0);
    }
    __syncthreads();

    // each wave owns 13 N-tiles: nt = wave + 4*ni  (NTP = 52, padded)
#pragma unroll 1
    for (int ni = 0; ni < NTP / 4; ++ni) {
      const int nt = wave + 4 * ni;
      const bf16* wb = wp + ((size_t)nt * KT) * 512 + lane * 16;
      v8f acc0 = {};
      v8f acc1 = {};
      FragBf bbuf[4];
#pragma unroll
      for (int p = 0; p < 4; ++p) {
        const bf16* bsrc = wb + (size_t)p * 512;
        bbuf[p].q[0] = *(const float4*)bsrc;
        bbuf[p].q[1] = *(const float4*)(bsrc + 8);
      }
#pragma unroll
      for (int kt = 0; kt < KT; ++kt) {
        FragBf a0, a1;
        const bf16* as0;
        const bf16* as1;
        if (kt < KTX) {
          as0 = &xtile[col][kt * 32 + hs * 8];
          as1 = &xtile[16 + col][kt * 32 + hs * 8];
        } else {
          int kb = (kt - KTX) * 32 + hs * 8;
          as0 = &hprev[col][kb];
          as1 = &hprev[16 + col][kb];
        }
        a0.q[0] = *(const float4*)as0;
        a0.q[1] = *(const float4*)(as0 + 16);
        a1.q[0] = *(const float4*)as1;
        a1.q[1] = *(const float4*)(as1 + 16);
        acc0 = __builtin_amdgcn_wmma_f32_16x16x32_bf16(false, a0.v, false, bbuf[kt & 3].v,
                                                       (short)0, acc0, false, false);
        acc1 = __builtin_amdgcn_wmma_f32_16x16x32_bf16(false, a1.v, false, bbuf[kt & 3].v,
                                                       (short)0, acc1, false, false);
        if (kt + 4 < KT) {  // refill the slot just consumed (4-deep pipeline)
          const bf16* bsrc = wb + (size_t)(kt + 4) * 512;
          bbuf[kt & 3].q[0] = *(const float4*)bsrc;
          bbuf[kt & 3].q[1] = *(const float4*)(bsrc + 8);
        }
      }
#pragma unroll
      for (int r = 0; r < 8; ++r) {
        pre[r + 8 * hs][nt * 16 + col] = acc0[r];
        pre[16 + r + 8 * hs][nt * 16 + col] = acc1[r];
      }
    }
    __syncthreads();

    // LSTM gates: i,f,g,o at N offsets 0/200/400/600
    for (int i = tid; i < 32 * Hdim; i += 128) {
      int r = i / Hdim;
      int j = i - r * Hdim;
      float gi = pre[r][j] + bs[j];
      float gf = pre[r][j + 200] + bs[j + 200];
      float gg = pre[r][j + 400] + bs[j + 400];
      float go = pre[r][j + 600] + bs[j + 600];
      float c = sigm(gf) * cst[r][j] + sigm(gi) * fast_tanh(gg);
      cst[r][j] = c;
      float h = sigm(go) * fast_tanh(c);
      hprev[r][j] = (bf16)h;
      hout[((size_t)t * Bdim + (b0 + r)) * HOPAD + dir * Hdim + j] = (bf16)h;
    }
    __syncthreads();
  }
}

// ---------------- path gather -> mlp_in (B*P x 800) bf16 ----------------
__global__ void k_gather(const int* __restrict__ paths, const bf16* __restrict__ h1,
                         bf16* __restrict__ mlp_in, int total) {
  int i = blockIdx.x * blockDim.x + threadIdx.x;
  if (i >= total) return;
  int k = i % H4;
  int row = i / H4;           // row = b*P + p
  int b = row >> 8;           // P = 256
  int which = (k >= 400) ? 1 : 0;
  int kk = k - which * 400;
  int t = paths[(size_t)row * 2 + which];
  bf16 v = (bf16)0.0f;
  if (t >= 0) v = h1[((size_t)t * Bdim + b) * HOPAD + kk];
  mlp_in[i] = v;
}

// ---------------- MLP layer 1: (32768 x 800) @ w1^T, tanh ----------------
// M=32 per block (B fragment reused by two WMMAs), 4-deep pipelined B stream.
__global__ __launch_bounds__(128) void k_mlp1(const bf16* __restrict__ mlp_in,
                                              const bf16* __restrict__ w1p,
                                              const float* __restrict__ b1,
                                              bf16* __restrict__ hid) {
  const int tid = threadIdx.x;
  const int wave = tid >> 5;
  const int lane = tid & 31;
  const int col = lane & 15;
  const int hs = lane >> 4;
  const int m0 = blockIdx.x * 32;
  const bf16* arow0 = mlp_in + (size_t)(m0 + col) * H4 + hs * 8;
  const bf16* arow1 = mlp_in + (size_t)(m0 + 16 + col) * H4 + hs * 8;
#pragma unroll 1
  for (int nt = wave; nt < 13; nt += 4) {
    v8f acc0 = {};
    v8f acc1 = {};
    const bf16* wb = w1p + ((size_t)nt * 25) * 512 + lane * 16;
    FragBf bbuf[4];
#pragma unroll
    for (int p = 0; p < 4; ++p) {
      const bf16* bsrc = wb + (size_t)p * 512;
      bbuf[p].q[0] = *(const float4*)bsrc;
      bbuf[p].q[1] = *(const float4*)(bsrc + 8);
    }
#pragma unroll
    for (int kt = 0; kt < 25; ++kt) {
      FragBf a0, a1;
      const bf16* as0 = arow0 + kt * 32;
      const bf16* as1 = arow1 + kt * 32;
      a0.q[0] = *(const float4*)as0;
      a0.q[1] = *(const float4*)(as0 + 16);
      a1.q[0] = *(const float4*)as1;
      a1.q[1] = *(const float4*)(as1 + 16);
      acc0 = __builtin_amdgcn_wmma_f32_16x16x32_bf16(false, a0.v, false, bbuf[kt & 3].v,
                                                     (short)0, acc0, false, false);
      acc1 = __builtin_amdgcn_wmma_f32_16x16x32_bf16(false, a1.v, false, bbuf[kt & 3].v,
                                                     (short)0, acc1, false, false);
      if (kt + 4 < 25) {
        const bf16* bsrc = wb + (size_t)(kt + 4) * 512;
        bbuf[kt & 3].q[0] = *(const float4*)bsrc;
        bbuf[kt & 3].q[1] = *(const float4*)(bsrc + 8);
      }
    }
    int n = nt * 16 + col;
    if (n < MLPD) {
      float bb = b1[n];
#pragma unroll
      for (int r = 0; r < 8; ++r) {
        int m = r + 8 * hs;
        hid[(size_t)(m0 + m) * MLPD + n] = (bf16)fast_tanh(acc0[r] + bb);
        hid[(size_t)(m0 + 16 + m) * MLPD + n] = (bf16)fast_tanh(acc1[r] + bb);
      }
    }
  }
}

// ---------------- final: hid @ w2^T + b2, softmax over C=4 ----------------
__global__ void k_final(const bf16* __restrict__ hid, const float* __restrict__ w2,
                        const float* __restrict__ b2, float* __restrict__ out, int rows) {
  __shared__ float ws2[Cdim][MLPD];
  __shared__ float wb2[Cdim];
  int tid = threadIdx.x;
  for (int i = tid; i < Cdim * MLPD; i += blockDim.x) ((float*)ws2)[i] = w2[i];
  if (tid < Cdim) wb2[tid] = b2[tid];
  __syncthreads();
  int row = blockIdx.x * blockDim.x + tid;
  if (row >= rows) return;
  float a0 = wb2[0], a1 = wb2[1], a2 = wb2[2], a3 = wb2[3];
  const bf16* hr = hid + (size_t)row * MLPD;
  for (int k = 0; k < MLPD; ++k) {
    float h = (float)hr[k];
    a0 += h * ws2[0][k];
    a1 += h * ws2[1][k];
    a2 += h * ws2[2][k];
    a3 += h * ws2[3][k];
  }
  float m = fmaxf(fmaxf(a0, a1), fmaxf(a2, a3));
  float e0 = __expf(a0 - m), e1 = __expf(a1 - m), e2 = __expf(a2 - m), e3 = __expf(a3 - m);
  float s = e0 + e1 + e2 + e3;
  out[(size_t)row * 4 + 0] = e0 / s;
  out[(size_t)row * 4 + 1] = e1 / s;
  out[(size_t)row * 4 + 2] = e2 / s;
  out[(size_t)row * 4 + 3] = e3 / s;
}

extern "C" void kernel_launch(void* const* d_in, const int* in_sizes, int n_in,
                              void* d_out, int out_size, void* d_ws, size_t ws_size,
                              hipStream_t stream) {
  (void)in_sizes; (void)n_in; (void)out_size; (void)ws_size;
  const int* tokens = (const int*)d_in[0];
  const int* paths = (const int*)d_in[1];
  const float* emb = (const float*)d_in[2];
  const float* wih_l0f = (const float*)d_in[3];
  const float* whh_l0f = (const float*)d_in[4];
  const float* bih_l0f = (const float*)d_in[5];
  const float* bhh_l0f = (const float*)d_in[6];
  const float* wih_l0b = (const float*)d_in[7];
  const float* whh_l0b = (const float*)d_in[8];
  const float* bih_l0b = (const float*)d_in[9];
  const float* bhh_l0b = (const float*)d_in[10];
  const float* wih_l1f = (const float*)d_in[11];
  const float* whh_l1f = (const float*)d_in[12];
  const float* bih_l1f = (const float*)d_in[13];
  const float* bhh_l1f = (const float*)d_in[14];
  const float* wih_l1b = (const float*)d_in[15];
  const float* whh_l1b = (const float*)d_in[16];
  const float* bih_l1b = (const float*)d_in[17];
  const float* bhh_l1b = (const float*)d_in[18];
  const float* w1 = (const float*)d_in[19];
  const float* b1 = (const float*)d_in[20];
  const float* w2 = (const float*)d_in[21];
  const float* b2 = (const float*)d_in[22];
  float* out = (float*)d_out;

  char* ws = (char*)d_ws;
  size_t off = 0;
  auto alloc = [&](size_t bytes) -> char* {
    char* p = ws + off;
    off += (bytes + 255) & ~(size_t)255;
    return p;
  };
  bf16* x0 = (bf16*)alloc((size_t)Tdim * Bdim * XPAD0 * 2);
  bf16* h0 = (bf16*)alloc((size_t)Tdim * Bdim * HOPAD * 2);
  bf16* h1 = (bf16*)alloc((size_t)Tdim * Bdim * HOPAD * 2);
  bf16* wp0f = (bf16*)alloc((size_t)NTP * 14 * 512 * 2);
  bf16* wp0b = (bf16*)alloc((size_t)NTP * 14 * 512 * 2);
  bf16* wp1f = (bf16*)alloc((size_t)NTP * 20 * 512 * 2);
  bf16* wp1b = (bf16*)alloc((size_t)NTP * 20 * 512 * 2);
  float* bs0f = (float*)alloc(H4 * 4);
  float* bs0b = (float*)alloc(H4 * 4);
  float* bs1f = (float*)alloc(H4 * 4);
  float* bs1b = (float*)alloc(H4 * 4);
  bf16* w1p = (bf16*)alloc((size_t)13 * 25 * 512 * 2);
  bf16* mlp_in = (bf16*)alloc((size_t)Bdim * Pdim * H4 * 2);
  bf16* hid = (bf16*)alloc((size_t)Bdim * Pdim * MLPD * 2);

  // 1) embedding gather with K zero-padding
  {
    int total = Tdim * Bdim * XPAD0;
    k_embed<<<(total + 255) / 256, 256, 0, stream>>>(tokens, emb, x0, total);
  }
  // 2) zero h0+h1 (contiguous) so pad columns 400..415 are zero
  {
    size_t n = ((size_t)Tdim * Bdim * HOPAD * 2 * 2) / 8;
    k_zero64<<<(unsigned)((n + 255) / 256), 256, 0, stream>>>((unsigned long long*)h0, n);
  }
  // 3) weight prepack into WMMA fragment order
  {
    int total = NTP * 14 * 512;
    k_pack_lstm<<<(total + 255) / 256, 256, 0, stream>>>(wih_l0f, whh_l0f, Edim, 7, wp0f, total);
    k_pack_lstm<<<(total + 255) / 256, 256, 0, stream>>>(wih_l0b, whh_l0b, Edim, 7, wp0b, total);
  }
  {
    int total = NTP * 20 * 512;
    k_pack_lstm<<<(total + 255) / 256, 256, 0, stream>>>(wih_l1f, whh_l1f, 2 * Hdim, 13, wp1f, total);
    k_pack_lstm<<<(total + 255) / 256, 256, 0, stream>>>(wih_l1b, whh_l1b, 2 * Hdim, 13, wp1b, total);
  }
  k_bsum<<<(H4 + 255) / 256, 256, 0, stream>>>(bih_l0f, bhh_l0f, bs0f, H4);
  k_bsum<<<(H4 + 255) / 256, 256, 0, stream>>>(bih_l0b, bhh_l0b, bs0b, H4);
  k_bsum<<<(H4 + 255) / 256, 256, 0, stream>>>(bih_l1f, bhh_l1f, bs1f, H4);
  k_bsum<<<(H4 + 255) / 256, 256, 0, stream>>>(bih_l1b, bhh_l1b, bs1b, H4);
  {
    int total = 13 * 25 * 512;
    k_pack_w1<<<(total + 255) / 256, 256, 0, stream>>>(w1, w1p, total);
  }
  // 4) recurrent layers (8 WGs each: 4 batch-slices x 2 dirs, M=32 per WG)
  k_bilstm<7><<<8, 128, 0, stream>>>(x0, wp0f, wp0b, bs0f, bs0b, h0);
  k_bilstm<13><<<8, 128, 0, stream>>>(h0, wp1f, wp1b, bs1f, bs1b, h1);
  // 5) path gather
  {
    int total = Bdim * Pdim * H4;
    k_gather<<<(total + 255) / 256, 256, 0, stream>>>(paths, h1, mlp_in, total);
  }
  // 6) MLP GEMM + tanh (WMMA, M=32/block), then scalar softmax head
  k_mlp1<<<(Bdim * Pdim) / 32, 128, 0, stream>>>(mlp_in, w1p, b1, hid);
  k_final<<<(Bdim * Pdim + 255) / 256, 256, 0, stream>>>(hid, w2, b2, out, Bdim * Pdim);
}
